// GeoGNNModel_53601191854193
// MI455X (gfx1250) — matrix operand order repaired and checked by reading, not compiled, not run on hardware
//
#include <hip/hip_runtime.h>

typedef __attribute__((ext_vector_type(16))) __bf16 v16bf;
typedef __attribute__((ext_vector_type(8)))  float  v8f;

union Frag { v16bf v; unsigned int u[8]; };

__device__ __forceinline__ unsigned short f2bf(float f) {
  unsigned int u = __float_as_uint(f);
  u += 0x7FFFu + ((u >> 16) & 1u);     // round-to-nearest-even truncation
  return (unsigned short)(u >> 16);
}

// ---------------------------------------------------------------------------
// Weight convert: W [K,Nc] f32 row-major  ->  Wt [Nc,K] bf16 (N-major)
// ---------------------------------------------------------------------------
__global__ void k_w2bf_t(const float* __restrict__ W, unsigned short* __restrict__ Wt,
                         int K, int Nc) {
  int i = blockIdx.x * blockDim.x + threadIdx.x;
  if (i >= K * Nc) return;
  int k = i / Nc, n = i % Nc;
  Wt[(size_t)n * K + k] = f2bf(W[i]);
}

// ---------------------------------------------------------------------------
// Atom embedding, one wave32 per node: lane k holds RBF weight for center k
// (20 centers, step 0.1), broadcast via __shfl.
// ---------------------------------------------------------------------------
__global__ void k_embed_atom(const float* __restrict__ x,
    const float* __restrict__ t0, const float* __restrict__ t1,
    const float* __restrict__ t2, const float* __restrict__ t3,
    const float* __restrict__ t4, const float* __restrict__ t5,
    const float* __restrict__ t6, const float* __restrict__ t7,
    const float* __restrict__ t8,
    const float* __restrict__ rbfW, const float* __restrict__ rbfb,
    float* __restrict__ out, int M) {
  int wave = threadIdx.x >> 5, lane = threadIdx.x & 31;
  int node = blockIdx.x * 8 + wave;
  if (node >= M) return;
  const float* r = x + (size_t)node * 10;
  int i0 = (int)r[0], i1 = (int)r[1], i2 = (int)r[2], i3 = (int)r[3];
  int i4 = (int)r[4], i5 = (int)r[5], i6 = (int)r[6], i7 = (int)r[7];
  int i8 = (int)r[8];
  float tv = r[9];
  float d = tv - 0.1f * (float)lane;
  float rbfv = (lane < 20) ? __expf(-10.f * d * d) : 0.f;
  #pragma unroll
  for (int chunk = 0; chunk < 8; ++chunk) {
    int c = chunk * 32 + lane;
    float h = rbfb[c];
    h += t0[i0 * 256 + c] + t1[i1 * 256 + c] + t2[i2 * 256 + c];
    h += t3[i3 * 256 + c] + t4[i4 * 256 + c] + t5[i5 * 256 + c];
    h += t6[i6 * 256 + c] + t7[i7 * 256 + c] + t8[i8 * 256 + c];
    #pragma unroll
    for (int k = 0; k < 20; ++k)
      h = fmaf(__shfl(rbfv, k, 32), rbfW[k * 256 + c], h);
    out[(size_t)node * 256 + c] = h;
  }
}

// ---------------------------------------------------------------------------
// Bond embedding, one wave32 per row: 3 int tables + RBF(20 centers)
// ---------------------------------------------------------------------------
__global__ void k_embed_bond(const float* __restrict__ x,
    const float* __restrict__ t0, const float* __restrict__ t1,
    const float* __restrict__ t2,
    const float* __restrict__ rbfW, const float* __restrict__ rbfb,
    float* __restrict__ out, int M) {
  int wave = threadIdx.x >> 5, lane = threadIdx.x & 31;
  int node = blockIdx.x * 8 + wave;
  if (node >= M) return;
  const float* r = x + (size_t)node * 4;
  int i0 = (int)r[0], i1 = (int)r[1], i2 = (int)r[2];
  float tv = r[3];
  float d = tv - 0.1f * (float)lane;
  float rbfv = (lane < 20) ? __expf(-10.f * d * d) : 0.f;
  #pragma unroll
  for (int chunk = 0; chunk < 8; ++chunk) {
    int c = chunk * 32 + lane;
    float h = rbfb[c];
    h += t0[i0 * 256 + c] + t1[i1 * 256 + c] + t2[i2 * 256 + c];
    #pragma unroll
    for (int k = 0; k < 20; ++k)
      h = fmaf(__shfl(rbfv, k, 32), rbfW[k * 256 + c], h);
    out[(size_t)node * 256 + c] = h;
  }
}

// ---------------------------------------------------------------------------
// Atom-graph scatter: msg = relu(x[src] + e); atomicAdd into agg[dst]
// ---------------------------------------------------------------------------
__global__ void k_scatter_atom(const float* __restrict__ xh,
                               const float* __restrict__ eh,
                               const int* __restrict__ ei,
                               float* __restrict__ agg, int E) {
  int i = blockIdx.x * 256 + threadIdx.x;
  if (i >= E * 64) return;
  int e = i >> 6, c = (i & 63) * 4;
  int src = ei[e], dst = ei[E + e];
  float4 xv = *(const float4*)&xh[(size_t)src * 256 + c];
  float4 ev = *(const float4*)&eh[(size_t)e * 256 + c];
  float* a = &agg[(size_t)dst * 256 + c];
  atomicAdd(a + 0, fmaxf(xv.x + ev.x, 0.f));
  atomicAdd(a + 1, fmaxf(xv.y + ev.y, 0.f));
  atomicAdd(a + 2, fmaxf(xv.z + ev.z, 0.f));
  atomicAdd(a + 3, fmaxf(xv.w + ev.w, 0.f));
}

// ---------------------------------------------------------------------------
// Bond-graph scatter with on-the-fly angle RBF (32 centers, step 0.1).
// ---------------------------------------------------------------------------
__global__ void k_scatter_bond(const float* __restrict__ xh,
                               const float* __restrict__ ang,
                               const int* __restrict__ ei,
                               const float* __restrict__ aW,
                               const float* __restrict__ ab,
                               float* __restrict__ agg, int EB) {
  int wave = threadIdx.x >> 5, lane = threadIdx.x & 31;
  int e = blockIdx.x * 8 + wave;
  if (e >= EB) return;
  float av = ang[e];
  float d = av - 0.1f * (float)lane;
  float rbfv = __expf(-10.f * d * d);
  int src = ei[e], dst = ei[EB + e];
  #pragma unroll
  for (int chunk = 0; chunk < 8; ++chunk) {
    int c = chunk * 32 + lane;
    float v = ab[c];
    #pragma unroll
    for (int k = 0; k < 32; ++k)
      v = fmaf(__shfl(rbfv, k, 32), aW[k * 256 + c], v);
    float msg = fmaxf(xh[(size_t)src * 256 + c] + v, 0.f);
    atomicAdd(&agg[(size_t)dst * 256 + c], msg);
  }
}

// ---------------------------------------------------------------------------
// Fused GIN MLP block, M-tile = 64 (4 row-subtiles of 16 kept in registers):
//   h0 = x + agg (bf16 tile in LDS)
//   H  = relu(h0 @ W1 + b1)   (WMMA bf16, each B frag feeds 4 WMMAs)
//   D  = H @ W2 + b2          (WMMA bf16)
//   out = LN(D)*g+b [, relu] + x   (in-place)
// LDS: Abuf 32K | Hbuf 64K ; Dbuf (64K f32) aliased over [0,64K) after barrier.
// ---------------------------------------------------------------------------
__global__ void __launch_bounds__(256) k_mlp_gin(
    const float* __restrict__ x, const float* __restrict__ agg,
    const unsigned short* __restrict__ W1t, const float* __restrict__ b1,
    const unsigned short* __restrict__ W2t, const float* __restrict__ b2,
    const float* __restrict__ lng, const float* __restrict__ lnb,
    float* __restrict__ out, int M, int last_act) {
  __shared__ char smem[98304];                                   // 96 KB
  unsigned short (*Abuf)[256] = (unsigned short (*)[256])smem;           // 32 KB
  unsigned short (*Hbuf)[512] = (unsigned short (*)[512])(smem + 32768); // 64 KB
  float          (*Dbuf)[256] = (float (*)[256])smem;                    // 64 KB (aliased)

  const int tid  = threadIdx.x;
  const int wave = tid >> 5, lane = tid & 31;
  const int lrow = lane & 15, grp = lane >> 4;
  const size_t rowBase = (size_t)blockIdx.x * 64;
  const v8f zero = {0.f, 0.f, 0.f, 0.f, 0.f, 0.f, 0.f, 0.f};

  // Phase 0: stage h0 = x + agg as bf16 (float4 granularity)
  for (int i = tid; i < 64 * 64; i += 256) {
    int rr = i >> 6, c4 = (i & 63) << 2;
    size_t gr = rowBase + rr;
    float4 xv = {0.f, 0.f, 0.f, 0.f}, av = {0.f, 0.f, 0.f, 0.f};
    if (gr < (size_t)M) {
      xv = *(const float4*)&x[gr * 256 + c4];
      av = *(const float4*)&agg[gr * 256 + c4];
    }
    unsigned int lo = (unsigned int)f2bf(xv.x + av.x) | ((unsigned int)f2bf(xv.y + av.y) << 16);
    unsigned int hi = (unsigned int)f2bf(xv.z + av.z) | ((unsigned int)f2bf(xv.w + av.w) << 16);
    uint2 pk = {lo, hi};
    *(uint2*)&Abuf[rr][c4] = pk;
  }
  __syncthreads();

  // Phase 1: GEMM1  (64 x 256) x (256 x 512); per wave: 4 N-tiles x 4 row-subtiles
  v8f acc1[4][4];
  #pragma unroll
  for (int rs = 0; rs < 4; ++rs)
    #pragma unroll
    for (int t = 0; t < 4; ++t) acc1[rs][t] = zero;

  for (int kt = 0; kt < 8; ++kt) {
    const int kb = kt * 32;
    Frag a[4], b[4];
    #pragma unroll
    for (int rs = 0; rs < 4; ++rs)
      #pragma unroll
      for (int p = 0; p < 8; ++p) {
        int kk = kb + ((p < 4) ? (grp * 8 + 2 * p) : (16 + grp * 8 + 2 * (p - 4)));
        a[rs].u[p] = *(const unsigned int*)&Abuf[rs * 16 + lrow][kk];
      }
    #pragma unroll
    for (int t = 0; t < 4; ++t) {
      int ncol = (wave * 4 + t) * 16 + lrow;
      #pragma unroll
      for (int p = 0; p < 8; ++p) {
        int kk = kb + ((p < 4) ? (grp * 8 + 2 * p) : (16 + grp * 8 + 2 * (p - 4)));
        b[t].u[p] = *(const unsigned int*)&W1t[(size_t)ncol * 256 + kk];
      }
    }
    #pragma unroll
    for (int t = 0; t < 4; ++t)
      #pragma unroll
      for (int rs = 0; rs < 4; ++rs)
        acc1[rs][t] = __builtin_amdgcn_wmma_f32_16x16x32_bf16(
            false, a[rs].v, false, b[t].v, (short)0, acc1[rs][t], false, false);
  }
  #pragma unroll
  for (int t = 0; t < 4; ++t) {
    int ncol = (wave * 4 + t) * 16 + lrow;
    float bias = b1[ncol];
    #pragma unroll
    for (int rs = 0; rs < 4; ++rs)
      #pragma unroll
      for (int vv = 0; vv < 8; ++vv)
        Hbuf[rs * 16 + grp * 8 + vv][ncol] = f2bf(fmaxf(acc1[rs][t][vv] + bias, 0.f));
  }
  __syncthreads();

  // Phase 2: GEMM2  (64 x 512) x (512 x 256); per wave: 2 N-tiles x 4 row-subtiles
  v8f acc2[4][2];
  #pragma unroll
  for (int rs = 0; rs < 4; ++rs) { acc2[rs][0] = zero; acc2[rs][1] = zero; }

  for (int kt = 0; kt < 16; ++kt) {
    const int kb = kt * 32;
    Frag a[4], b[2];
    #pragma unroll
    for (int rs = 0; rs < 4; ++rs)
      #pragma unroll
      for (int p = 0; p < 8; ++p) {
        int kk = kb + ((p < 4) ? (grp * 8 + 2 * p) : (16 + grp * 8 + 2 * (p - 4)));
        a[rs].u[p] = *(const unsigned int*)&Hbuf[rs * 16 + lrow][kk];
      }
    #pragma unroll
    for (int t = 0; t < 2; ++t) {
      int ncol = (wave * 2 + t) * 16 + lrow;
      #pragma unroll
      for (int p = 0; p < 8; ++p) {
        int kk = kb + ((p < 4) ? (grp * 8 + 2 * p) : (16 + grp * 8 + 2 * (p - 4)));
        b[t].u[p] = *(const unsigned int*)&W2t[(size_t)ncol * 512 + kk];
      }
    }
    #pragma unroll
    for (int t = 0; t < 2; ++t)
      #pragma unroll
      for (int rs = 0; rs < 4; ++rs)
        acc2[rs][t] = __builtin_amdgcn_wmma_f32_16x16x32_bf16(
            false, a[rs].v, false, b[t].v, (short)0, acc2[rs][t], false, false);
  }
  __syncthreads();   // Hbuf/Abuf now dead; Dbuf may alias them

  #pragma unroll
  for (int t = 0; t < 2; ++t) {
    int ncol = (wave * 2 + t) * 16 + lrow;
    float bias = b2[ncol];
    #pragma unroll
    for (int rs = 0; rs < 4; ++rs)
      #pragma unroll
      for (int vv = 0; vv < 8; ++vv)
        Dbuf[rs * 16 + grp * 8 + vv][ncol] = acc2[rs][t][vv] + bias;
  }
  __syncthreads();

  // Phase 3: LayerNorm + optional ReLU + residual; 8 rows per wave
  #pragma unroll
  for (int rr = 0; rr < 8; ++rr) {
    int r = wave * 8 + rr;
    float s = 0.f, ss = 0.f;
    #pragma unroll
    for (int j = 0; j < 8; ++j) {
      float v = Dbuf[r][lane + 32 * j];
      s += v; ss += v * v;
    }
    #pragma unroll
    for (int off = 16; off > 0; off >>= 1) {
      s  += __shfl_xor(s,  off, 32);
      ss += __shfl_xor(ss, off, 32);
    }
    float mean = s * (1.f / 256.f);
    float var  = ss * (1.f / 256.f) - mean * mean;
    float rstd = rsqrtf(var + 1e-5f);
    size_t gr = rowBase + r;
    if (gr < (size_t)M) {
      #pragma unroll
      for (int j = 0; j < 8; ++j) {
        int c = lane + 32 * j;
        float v = (Dbuf[r][c] - mean) * rstd * lng[c] + lnb[c];
        if (last_act) v = fmaxf(v, 0.f);
        out[gr * 256 + c] = v + x[gr * 256 + c];
      }
    }
  }
}

// ---------------------------------------------------------------------------
// Readout
// ---------------------------------------------------------------------------
__global__ void k_gsum(const float* __restrict__ xh, const int* __restrict__ batch,
                       float* __restrict__ gsum, int N) {
  int i = blockIdx.x * 256 + threadIdx.x;
  if (i >= N * 256) return;
  int node = i >> 8, c = i & 255;
  atomicAdd(&gsum[(size_t)batch[node] * 256 + c], xh[i]);
}
__global__ void k_gcount(const int* __restrict__ batch, float* __restrict__ counts, int N) {
  int i = blockIdx.x * 256 + threadIdx.x;
  if (i >= N) return;
  atomicAdd(&counts[batch[i]], 1.f);
}
__global__ void k_gdiv(float* __restrict__ g, const float* __restrict__ counts, int G) {
  int i = blockIdx.x * 256 + threadIdx.x;
  if (i >= G * 256) return;
  g[i] = g[i] / fmaxf(counts[i >> 8], 1.f);
}

// ---------------------------------------------------------------------------
extern "C" void kernel_launch(void* const* d_in, const int* in_sizes, int n_in,
                              void* d_out, int out_size, void* d_ws, size_t ws_size,
                              hipStream_t stream) {
  const float* x_atom  = (const float*)d_in[0];
  const float* ea_atom = (const float*)d_in[1];
  const float* x_bond  = (const float*)d_in[2];
  const float* angles  = (const float*)d_in[3];
  const int*   ei_atom = (const int*)d_in[4];
  const int*   ei_bond = (const int*)d_in[5];
  const int*   batchA  = (const int*)d_in[6];
  // d_in[7] = x_bond_batch (unused by the reference outputs)

  const int N  = in_sizes[6];
  const int NB = in_sizes[7];
  const int E  = in_sizes[4] / 2;
  const int EB = in_sizes[5] / 2;
  const int G  = out_size / 256 - N - NB;

  // ---- params (insertion-order flatten of nested dict) ----
  int p = 8;
  const float* atom_tab[9];
  for (int i = 0; i < 9; ++i) atom_tab[i] = (const float*)d_in[p++];
  const float* atom_rbfW = (const float*)d_in[p++];
  const float* atom_rbfb = (const float*)d_in[p++];
  const float* bond_tab0[3];
  for (int i = 0; i < 3; ++i) bond_tab0[i] = (const float*)d_in[p++];
  const float* bond_rbfW0 = (const float*)d_in[p++];
  const float* bond_rbfb0 = (const float*)d_in[p++];

  struct Layer {
    const float *bt0, *bt1, *bt2, *brW, *brb, *aW, *ab;
    const float *m1W, *m1b, *m2W, *m2b, *lg, *lb;   // ab block
    const float *n1W, *n1b, *n2W, *n2b, *mg, *mb;   // ba block
  } Lp[5];
  for (int l = 0; l < 5; ++l) {
    Lp[l].bt0 = (const float*)d_in[p++];
    Lp[l].bt1 = (const float*)d_in[p++];
    Lp[l].bt2 = (const float*)d_in[p++];
    Lp[l].brW = (const float*)d_in[p++];
    Lp[l].brb = (const float*)d_in[p++];
    Lp[l].aW  = (const float*)d_in[p++];
    Lp[l].ab  = (const float*)d_in[p++];
    Lp[l].m1W = (const float*)d_in[p++];
    Lp[l].m1b = (const float*)d_in[p++];
    Lp[l].m2W = (const float*)d_in[p++];
    Lp[l].m2b = (const float*)d_in[p++];
    Lp[l].lg  = (const float*)d_in[p++];
    Lp[l].lb  = (const float*)d_in[p++];
    Lp[l].n1W = (const float*)d_in[p++];
    Lp[l].n1b = (const float*)d_in[p++];
    Lp[l].n2W = (const float*)d_in[p++];
    Lp[l].n2b = (const float*)d_in[p++];
    Lp[l].mg  = (const float*)d_in[p++];
    Lp[l].mb  = (const float*)d_in[p++];
  }

  // ---- workspace layout ----
  char* ws = (char*)d_ws;
  int maxRows = (NB > N) ? NB : N;
  float* agg = (float*)ws;
  size_t aggBytes = (size_t)maxRows * 256 * sizeof(float);
  unsigned short* wbf = (unsigned short*)(ws + aggBytes);
  auto wl = [&](int l, int which) -> unsigned short* {
    return wbf + (size_t)l * 524288 + (size_t)which * 131072;
  };
  float* counts = (float*)(ws + aggBytes + (size_t)5 * 1048576);

  // ---- output regions (node_h / edge_h updated in-place) ----
  float* node_h = (float*)d_out;
  float* edge_h = node_h + (size_t)N * 256;
  float* graph  = edge_h + (size_t)NB * 256;

  // ---- weight conversion (bf16, N-major transpose) ----
  for (int l = 0; l < 5; ++l) {
    k_w2bf_t<<<(256 * 512 + 255) / 256, 256, 0, stream>>>(Lp[l].m1W, wl(l, 0), 256, 512);
    k_w2bf_t<<<(512 * 256 + 255) / 256, 256, 0, stream>>>(Lp[l].m2W, wl(l, 1), 512, 256);
    k_w2bf_t<<<(256 * 512 + 255) / 256, 256, 0, stream>>>(Lp[l].n1W, wl(l, 2), 256, 512);
    k_w2bf_t<<<(512 * 256 + 255) / 256, 256, 0, stream>>>(Lp[l].n2W, wl(l, 3), 512, 256);
  }

  // ---- initial embeddings ----
  k_embed_atom<<<(N + 7) / 8, 256, 0, stream>>>(
      x_atom, atom_tab[0], atom_tab[1], atom_tab[2], atom_tab[3], atom_tab[4],
      atom_tab[5], atom_tab[6], atom_tab[7], atom_tab[8],
      atom_rbfW, atom_rbfb, node_h, N);
  k_embed_bond<<<(E + 7) / 8, 256, 0, stream>>>(
      ea_atom, bond_tab0[0], bond_tab0[1], bond_tab0[2],
      bond_rbfW0, bond_rbfb0, edge_h, E);

  // ---- layers ----
  for (int l = 0; l < 5; ++l) {
    int last_act = (l != 4) ? 1 : 0;
    // atom-graph GIN block
    hipMemsetAsync(agg, 0, (size_t)N * 256 * sizeof(float), stream);
    k_scatter_atom<<<(E * 64 + 255) / 256, 256, 0, stream>>>(node_h, edge_h, ei_atom, agg, E);
    k_mlp_gin<<<(N + 63) / 64, 256, 0, stream>>>(
        node_h, agg, wl(l, 0), Lp[l].m1b, wl(l, 1), Lp[l].m2b,
        Lp[l].lg, Lp[l].lb, node_h, N, last_act);
    // fresh bond-node features for this layer
    k_embed_bond<<<(NB + 7) / 8, 256, 0, stream>>>(
        x_bond, Lp[l].bt0, Lp[l].bt1, Lp[l].bt2, Lp[l].brW, Lp[l].brb, edge_h, NB);
    // bond-graph GIN block (angle RBF computed in-scatter)
    hipMemsetAsync(agg, 0, (size_t)NB * 256 * sizeof(float), stream);
    k_scatter_bond<<<(EB + 7) / 8, 256, 0, stream>>>(
        edge_h, angles, ei_bond, Lp[l].aW, Lp[l].ab, agg, EB);
    k_mlp_gin<<<(NB + 63) / 64, 256, 0, stream>>>(
        edge_h, agg, wl(l, 2), Lp[l].n1b, wl(l, 3), Lp[l].n2b,
        Lp[l].mg, Lp[l].mb, edge_h, NB, last_act);
  }

  // ---- mean readout over atom graphs ----
  hipMemsetAsync(graph, 0, (size_t)G * 256 * sizeof(float), stream);
  hipMemsetAsync(counts, 0, (size_t)G * sizeof(float), stream);
  k_gsum<<<(N * 256 + 255) / 256, 256, 0, stream>>>(node_h, batchA, graph, N);
  k_gcount<<<(N + 255) / 256, 256, 0, stream>>>(batchA, counts, N);
  k_gdiv<<<(G * 256 + 255) / 256, 256, 0, stream>>>(graph, counts, G);
}
